// IntraSampleAttention_36661840839199
// MI455X (gfx1250) — matrix operand order repaired and checked by reading, not compiled, not run on hardware
//
#include <hip/hip_runtime.h>
#include <hip/hip_bf16.h>

// ---------------------------------------------------------------------------
// Mamba residual block for MI455X (gfx1250, wave32, WMMA bf16)
// B=8, C=256, H=W=32 (L=1024), D_IN=512, D_STATE=16, DT_RANK=16, D_CONV=3
// ---------------------------------------------------------------------------

typedef __attribute__((ext_vector_type(16))) __bf16 v16bf;
typedef __attribute__((ext_vector_type(8)))  float  v8f;

#define B_   8
#define C_   256
#define L_   1024
#define DIN_ 512
#define NST_ 16
#define M_TOT (B_ * L_)          // 8192

// ---------------- workspace layout (bytes) ----------------
static constexpr size_t OFF_NORMED = 0;                                   // bf16 8192x256
static constexpr size_t SZ_NORMED  = (size_t)M_TOT * C_ * 2;
static constexpr size_t OFF_XZ     = OFF_NORMED + SZ_NORMED;              // f32  8192x1024
static constexpr size_t SZ_XZ      = (size_t)M_TOT * 1024 * 4;
static constexpr size_t OFF_XC     = OFF_XZ + SZ_XZ;                      // bf16 8192x512
static constexpr size_t SZ_XC      = (size_t)M_TOT * DIN_ * 2;
static constexpr size_t OFF_XDBL   = OFF_XC + SZ_XC;                      // f32  8192x48
static constexpr size_t SZ_XDBL    = (size_t)M_TOT * 48 * 4;
static constexpr size_t OFF_DELTA  = OFF_XDBL + SZ_XDBL;                  // f32  8192x512
static constexpr size_t SZ_DELTA   = (size_t)M_TOT * DIN_ * 4;
static constexpr size_t OFF_YMOD   = OFF_DELTA + SZ_DELTA;                // bf16 8192x512

// ---------------------------------------------------------------------------
// Kernel 1: (B,C,H,W) -> RMSNorm over C -> normed bf16 (B, L, C)
// ---------------------------------------------------------------------------
__global__ __launch_bounds__(256)
void norm_transpose_kernel(const float* __restrict__ feat,
                           const float* __restrict__ norm_w,
                           __bf16* __restrict__ normed)
{
    __shared__ float tile[C_][33];     // [c][t], padded
    __shared__ float part[8][32];
    __shared__ float scale[32];

    const int b  = blockIdx.y;
    const int t0 = blockIdx.x * 32;
    const int tid = threadIdx.x;
    const int tx = tid & 31;           // t offset (coalesced reads along W)
    const int ty = tid >> 5;           // 0..7

    for (int cc = 0; cc < C_; cc += 8) {
        const int c = cc + ty;
        tile[c][tx] = feat[((size_t)b * C_ + c) * L_ + t0 + tx];
    }
    __syncthreads();

    {
        float ss = 0.f;
        const int t = tid & 31;
        const int cbase = (tid >> 5) * 32;
        #pragma unroll 8
        for (int c = 0; c < 32; ++c) {
            const float v = tile[cbase + c][t];
            ss += v * v;
        }
        part[tid >> 5][t] = ss;
    }
    __syncthreads();
    if (tid < 32) {
        float s = 0.f;
        #pragma unroll
        for (int j = 0; j < 8; ++j) s += part[j][tid];
        scale[tid] = rsqrtf(s * (1.0f / (float)C_) + 1e-5f);
    }
    __syncthreads();

    const int tt = tid >> 3;           // 0..31
    const int cb = (tid & 7) * 32;
    const float sc = scale[tt];
    const size_t ob = ((size_t)b * L_ + t0 + tt) * C_ + cb;
    #pragma unroll 8
    for (int c = 0; c < 32; ++c) {
        normed[ob + c] = (__bf16)(tile[cb + c][tt] * sc * norm_w[cb + c]);
    }
}

// ---------------------------------------------------------------------------
// WMMA GEMM: C[MxN] = A[MxK](bf16, row-major) * Bw[KxN](f32 -> bf16)
// Block tile 128x64, BK=32, 8 waves (4x2), each wave = 32Mx32N (4 v_wmma/step).
// A fragments loaded straight from global (two b128 per fragment: the ISA
// 16-bit A layout is two contiguous 8-element K-runs per lane).
// B tile staged transposed in LDS (Bs[n][k]) so each lane's fragment is a
// pair of 16B K-runs (as A) -> two ds_load_b128. Double-buffered, 1 barrier/step.
// ---------------------------------------------------------------------------
template<bool TRANS_RES, bool NGUARD>
__global__ __launch_bounds__(256)
void wmma_gemm_kernel(const __bf16* __restrict__ A,
                      const float* __restrict__ Bw,
                      float* __restrict__ Cout,
                      const float* __restrict__ resid,
                      int M, int N, int K)
{
    constexpr int BM = 128, BN = 64, BK = 32;
    __shared__ __bf16 Bs[2][BN][BK + 8];   // transposed tile, row stride 80B

    const int tid  = threadIdx.x;
    const int lane = tid & 31;
    const int wave = tid >> 5;
    const int hl   = lane >> 4;         // half-wave 0/1
    const int l16  = lane & 15;
    const int wm   = wave >> 1;         // 0..3, 32 rows each
    const int wn   = wave & 1;          // 0..1, 32 cols each
    const int m0   = blockIdx.y * BM;
    const int n0   = blockIdx.x * BN;

    // B-tile loader: each thread owns (col = tid&63, rows brow..brow+7)
    const int bcol = tid & 63;
    const int brow = (tid >> 6) * 8;
    const int gn   = n0 + bcol;

    // per-lane A row pointers (fragment rows for this wave)
    const __bf16* arow0 = A + (size_t)(m0 + wm * 32 + l16) * K;
    const __bf16* arow1 = arow0 + (size_t)16 * K;

    const int KT = K / BK;

    union V16 { v16bf v; uint4 q[2]; };
    union P8  { uint4 q; __bf16 h[8]; };

    float bv[8];
    {   // prologue: tile 0 -> LDS buf 0
        #pragma unroll
        for (int r = 0; r < 8; ++r) {
            const float v = (!NGUARD || gn < N) ? Bw[(size_t)(brow + r) * N + gn] : 0.0f;
            bv[r] = v;
        }
        P8 p;
        #pragma unroll
        for (int r = 0; r < 8; ++r) p.h[r] = (__bf16)bv[r];
        *(uint4*)&Bs[0][bcol][brow] = p.q;
    }
    __syncthreads();

    v8f acc[2][2];
    acc[0][0] = (v8f){}; acc[0][1] = (v8f){};
    acc[1][0] = (v8f){}; acc[1][1] = (v8f){};

    for (int kt = 0; kt < KT; ++kt) {
        const int cur = kt & 1;
        const int k0  = kt * BK;

        // issue next B tile's global loads before the math
        float bnext[8];
        if (kt + 1 < KT) {
            const int kb = k0 + BK + brow;
            #pragma unroll
            for (int r = 0; r < 8; ++r) {
                bnext[r] = (!NGUARD || gn < N) ? Bw[(size_t)(kb + r) * N + gn] : 0.0f;
            }
            // WGP-scope speculative prefetch of the A tile after next
            __builtin_prefetch(arow0 + k0 + 2 * BK, 0, 3);
        }

        // A fragments: two contiguous b128 runs per lane, direct from global
        V16 a0, a1;
        a0.q[0] = *(const uint4*)(arow0 + k0 + hl * 8);
        a0.q[1] = *(const uint4*)(arow0 + k0 + 16 + hl * 8);
        a1.q[0] = *(const uint4*)(arow1 + k0 + hl * 8);
        a1.q[1] = *(const uint4*)(arow1 + k0 + 16 + hl * 8);

        // B fragments: two 8-element K-runs per lane (same as A) from transposed LDS tile
        V16 b0, b1;
        const int nb = wn * 32 + l16;
        b0.q[0] = *(const uint4*)&Bs[cur][nb][hl * 8];
        b0.q[1] = *(const uint4*)&Bs[cur][nb][16 + hl * 8];
        b1.q[0] = *(const uint4*)&Bs[cur][nb + 16][hl * 8];
        b1.q[1] = *(const uint4*)&Bs[cur][nb + 16][16 + hl * 8];

        acc[0][0] = __builtin_amdgcn_wmma_f32_16x16x32_bf16(false, a0.v, false, b0.v,
                                                            (short)0, acc[0][0], false, false);
        acc[0][1] = __builtin_amdgcn_wmma_f32_16x16x32_bf16(false, a0.v, false, b1.v,
                                                            (short)0, acc[0][1], false, false);
        acc[1][0] = __builtin_amdgcn_wmma_f32_16x16x32_bf16(false, a1.v, false, b0.v,
                                                            (short)0, acc[1][0], false, false);
        acc[1][1] = __builtin_amdgcn_wmma_f32_16x16x32_bf16(false, a1.v, false, b1.v,
                                                            (short)0, acc[1][1], false, false);

        if (kt + 1 < KT) {
            P8 p;
            #pragma unroll
            for (int r = 0; r < 8; ++r) p.h[r] = (__bf16)bnext[r];
            *(uint4*)&Bs[cur ^ 1][bcol][brow] = p.q;
        }
        __syncthreads();
    }

    // epilogue: C/D layout -> VGPR r holds (M = hl*8 + r, N = l16)
    const int nc = n0 + wn * 32 + l16;
    if (!TRANS_RES) {
        #pragma unroll
        for (int sm = 0; sm < 2; ++sm) {
            const int mb = m0 + wm * 32 + sm * 16 + hl * 8;
            #pragma unroll
            for (int sn = 0; sn < 2; ++sn) {
                const int n = nc + sn * 16;
                if (!NGUARD || n < N) {
                    #pragma unroll
                    for (int r = 0; r < 8; ++r) {
                        Cout[(size_t)(mb + r) * N + n] = acc[sm][sn][r];
                    }
                }
            }
        }
    } else {
        // out (B, N, L): m = b*L + t ; each lane owns 8 contiguous t values
        // (32B aligned) -> residual add + store as b128 vectors.
        union F8 { v8f v; float4 f[2]; };
        const int mbase = m0 + wm * 32;
        const int bi = mbase >> 10;
        const int tb = (mbase & (L_ - 1)) + hl * 8;
        #pragma unroll
        for (int sm = 0; sm < 2; ++sm) {
            #pragma unroll
            for (int sn = 0; sn < 2; ++sn) {
                const size_t o = ((size_t)(bi * N + nc + sn * 16)) * L_ + tb + sm * 16;
                const float4* rp = (const float4*)(resid + o);
                float4* op = (float4*)(Cout + o);
                F8 u; u.v = acc[sm][sn];
                const float4 r0 = rp[0];
                const float4 r1 = rp[1];
                u.f[0].x += r0.x; u.f[0].y += r0.y; u.f[0].z += r0.z; u.f[0].w += r0.w;
                u.f[1].x += r1.x; u.f[1].y += r1.y; u.f[1].z += r1.z; u.f[1].w += r1.w;
                op[0] = u.f[0];
                op[1] = u.f[1];
            }
        }
    }
}

// ---------------------------------------------------------------------------
// Kernel 3: depthwise causal conv (k=3) along t + bias + SiLU -> xc bf16
// ---------------------------------------------------------------------------
__global__ __launch_bounds__(256)
void conv_silu_kernel(const float* __restrict__ xz,
                      const float* __restrict__ conv_w,
                      const float* __restrict__ conv_b,
                      __bf16* __restrict__ xc)
{
    const int idx = blockIdx.x * 256 + threadIdx.x;   // bt*512 + d
    const int d  = idx & (DIN_ - 1);
    const int bt = idx >> 9;
    const int t  = bt & (L_ - 1);

    float acc = conv_b[d];
    const float w0 = conv_w[d * 3 + 0];
    const float w1 = conv_w[d * 3 + 1];
    const float w2 = conv_w[d * 3 + 2];
    if (t >= 2) acc += xz[((size_t)bt - 2) * 1024 + d] * w0;
    if (t >= 1) acc += xz[((size_t)bt - 1) * 1024 + d] * w1;
    acc += xz[(size_t)bt * 1024 + d] * w2;
    const float s = acc / (1.0f + __expf(-acc));
    xc[idx] = (__bf16)s;
}

// ---------------------------------------------------------------------------
// Kernel 5: delta = softplus(x_dbl[:, :16] @ W_dt + b_dt)   (K=16 -> VALU)
// ---------------------------------------------------------------------------
__global__ __launch_bounds__(256)
void delta_kernel(const float* __restrict__ x_dbl,
                  const float* __restrict__ W_dt,
                  const float* __restrict__ b_dt,
                  float* __restrict__ delta)
{
    const int idx = blockIdx.x * 256 + threadIdx.x;   // bt*512 + d
    const int d  = idx & (DIN_ - 1);
    const int bt = idx >> 9;
    const float* xr = x_dbl + (size_t)bt * 48;
    float acc = b_dt[d];
    #pragma unroll
    for (int r = 0; r < 16; ++r) acc += xr[r] * W_dt[r * DIN_ + d];
    delta[idx] = (acc > 20.0f) ? acc : log1pf(__expf(acc));
}

// ---------------------------------------------------------------------------
// Kernel 6: selective scan. One thread per (d, n) state; 16-lane xor-shuffle
// reduction over n. Fuses y*silu(z).
// ---------------------------------------------------------------------------
__global__ __launch_bounds__(256)
void scan_kernel(const __bf16* __restrict__ xc,
                 const float* __restrict__ delta,
                 const float* __restrict__ x_dbl,
                 const float* __restrict__ xz,
                 const float* __restrict__ A_log,
                 const float* __restrict__ Dp,
                 __bf16* __restrict__ ymod)
{
    const int b    = blockIdx.x >> 5;        // 0..7
    const int dblk = blockIdx.x & 31;        // 16 d per block
    const int tid  = threadIdx.x;
    const int lane = tid & 31;
    const int wave = tid >> 5;
    const int n  = lane & 15;                // state index
    const int dl = lane >> 4;                // 0/1
    const int d  = dblk * 16 + wave * 2 + dl;

    const float Aval = -__expf(A_log[d * NST_ + n]);
    const float Dval = Dp[d];
    float s = 0.f;

    for (int t = 0; t < L_; ++t) {
        const int bt = b * L_ + t;
        const float dv = delta[(size_t)bt * DIN_ + d];
        const float u  = (float)xc[(size_t)bt * DIN_ + d];
        const float Bv = x_dbl[(size_t)bt * 48 + 16 + n];
        const float Cv = x_dbl[(size_t)bt * 48 + 32 + n];
        s = __expf(dv * Aval) * s + dv * Bv * u;
        float part = s * Cv;
        part += __shfl_xor(part, 8, 32);
        part += __shfl_xor(part, 4, 32);
        part += __shfl_xor(part, 2, 32);
        part += __shfl_xor(part, 1, 32);
        if (n == 0) {
            const float y  = part + u * Dval;
            const float zv = xz[(size_t)bt * 1024 + DIN_ + d];
            const float sz = zv / (1.0f + __expf(-zv));
            ymod[(size_t)bt * DIN_ + d] = (__bf16)(y * sz);
        }
    }
}

// ---------------------------------------------------------------------------
extern "C" void kernel_launch(void* const* d_in, const int* in_sizes, int n_in,
                              void* d_out, int out_size, void* d_ws, size_t ws_size,
                              hipStream_t stream)
{
    const float* feature = (const float*)d_in[0];
    const float* norm_w  = (const float*)d_in[1];
    const float* W_in    = (const float*)d_in[2];
    const float* conv_w  = (const float*)d_in[3];
    const float* conv_b  = (const float*)d_in[4];
    const float* W_x     = (const float*)d_in[5];
    const float* W_dt    = (const float*)d_in[6];
    const float* b_dt    = (const float*)d_in[7];
    const float* A_log   = (const float*)d_in[8];
    const float* Dp      = (const float*)d_in[9];
    const float* W_out   = (const float*)d_in[10];
    float* out = (float*)d_out;

    char* ws = (char*)d_ws;
    __bf16* normed = (__bf16*)(ws + OFF_NORMED);
    float*  xz     = (float*) (ws + OFF_XZ);
    __bf16* xc     = (__bf16*)(ws + OFF_XC);
    float*  x_dbl  = (float*) (ws + OFF_XDBL);
    float*  delta  = (float*) (ws + OFF_DELTA);
    __bf16* ymod   = (__bf16*)(ws + OFF_YMOD);

    // 1. transpose + RMSNorm
    norm_transpose_kernel<<<dim3(L_ / 32, B_), 256, 0, stream>>>(feature, norm_w, normed);

    // 2. xz = normed @ W_in   (8192 x 256 x 1024)
    wmma_gemm_kernel<false, false><<<dim3(1024 / 64, M_TOT / 128), 256, 0, stream>>>(
        normed, W_in, xz, nullptr, M_TOT, 1024, C_);

    // 3. depthwise conv + SiLU
    conv_silu_kernel<<<(M_TOT * DIN_) / 256, 256, 0, stream>>>(xz, conv_w, conv_b, xc);

    // 4. x_dbl = xc @ W_x     (8192 x 512 x 48)
    wmma_gemm_kernel<false, true><<<dim3(1, M_TOT / 128), 256, 0, stream>>>(
        xc, W_x, x_dbl, nullptr, M_TOT, 48, DIN_);

    // 5. delta = softplus(x_dbl[:, :16] @ W_dt + b_dt)
    delta_kernel<<<(M_TOT * DIN_) / 256, 256, 0, stream>>>(x_dbl, W_dt, b_dt, delta);

    // 6. selective scan + gating
    scan_kernel<<<B_ * (DIN_ / 16), 256, 0, stream>>>(xc, delta, x_dbl, xz, A_log, Dp, ymod);

    // 7. out = ymod @ W_out + residual, written directly in (B,C,H,W)
    wmma_gemm_kernel<true, false><<<dim3(C_ / 64, M_TOT / 128), 256, 0, stream>>>(
        ymod, W_out, out, feature, M_TOT, C_, DIN_);
}
